// attention_scaled_dot_product_3796751089884
// MI455X (gfx1250) — matrix operand order repaired and checked by reading, not compiled, not run on hardware
//
#include <hip/hip_runtime.h>

typedef _Float16 v16h __attribute__((ext_vector_type(16)));
typedef _Float16 h8   __attribute__((ext_vector_type(8)));
typedef _Float16 h4   __attribute__((ext_vector_type(4)));
typedef float    v8f  __attribute__((ext_vector_type(8)));

#define B_   16
#define NQ   2048
#define NK   2048
#define DD   128
#define NCH  (NK / 32)                    // 64 k-chunks of 32
#define LOG2E 1.4426950408889634f
#define SC2   (0.08838834764831845f * LOG2E)   // (1/sqrt(128)) * log2(e)
#define NEG_BIG2 (-1.0e10f * LOG2E)            // masked score in log2 domain

__global__ __launch_bounds__(256) void attn_fwd_kernel(
    const float* __restrict__ qg, const float* __restrict__ kg,
    const float* __restrict__ vg, const unsigned char* __restrict__ maskg,
    float* __restrict__ ctx, float* __restrict__ att)
{
    __shared__ _Float16 lds_k[32 * 128];    // K tile, [krow][d], fp16, 8KB
    __shared__ _Float16 lds_vt[128 * 32];   // V tile transposed, [d][krow], 8KB
    __shared__ _Float16 lds_p[8 * 16 * 32]; // per-wave P staging, 8KB

    const int tid   = threadIdx.x;          // 0..255, 8 waves
    const int w     = tid >> 5;
    const int lane  = tid & 31;
    const int ln    = lane & 15;
    const int g     = lane >> 4;            // half-wave group
    const int bb    = blockIdx.y;
    const int qrow0 = blockIdx.x * 128 + w * 16;

    // per-thread staging coordinates (4 float4 groups of the 32x128 tile)
    int srow[4], scol[4];
    #pragma unroll
    for (int i = 0; i < 4; ++i) {
        int f4  = tid + i * 256;
        srow[i] = f4 >> 5;
        scol[i] = (f4 & 31) * 4;
    }

    // ---------------- Q A-fragments (fp32 -> fp16), live in registers ------
    // A 16x32 f16 frag, contiguous k order chosen here: lane (ln,g): row = ln; halves j = 0..15 -> K = g*16 + j.
    v16h aq[4];
    {
        const float* qrow = qg + ((size_t)bb * NQ + qrow0 + ln) * DD;
        #pragma unroll
        for (int kkc = 0; kkc < 4; ++kkc) {
            const int b0 = kkc * 32 + g * 16;
            const int b1 = b0 + 8;
            float4 q0 = *(const float4*)(qrow + b0);
            float4 q1 = *(const float4*)(qrow + b0 + 4);
            float4 q2 = *(const float4*)(qrow + b1);
            float4 q3 = *(const float4*)(qrow + b1 + 4);
            v16h a = {(_Float16)q0.x, (_Float16)q0.y, (_Float16)q0.z, (_Float16)q0.w,
                      (_Float16)q1.x, (_Float16)q1.y, (_Float16)q1.z, (_Float16)q1.w,
                      (_Float16)q2.x, (_Float16)q2.y, (_Float16)q2.z, (_Float16)q2.w,
                      (_Float16)q3.x, (_Float16)q3.y, (_Float16)q3.z, (_Float16)q3.w};
            aq[kkc] = a;
        }
    }

    float m[8], l[8];
    #pragma unroll
    for (int r = 0; r < 8; ++r) { m[r] = -1.0e30f; l[r] = 0.0f; }

    // ================= PASS 1: online row max + exp2-sum ===================
    float4 kreg[4];
    #pragma unroll
    for (int i = 0; i < 4; ++i)     // preload chunk 0
        kreg[i] = *(const float4*)(kg + ((size_t)bb * NK + srow[i]) * DD + scol[i]);

    for (int kc = 0; kc < NCH; ++kc) {
        const int kr0 = kc * 32;
        __syncthreads();                    // prior chunk's LDS reads complete
        #pragma unroll
        for (int i = 0; i < 4; ++i) {       // cvt + store current chunk
            h4 hh = {(_Float16)kreg[i].x, (_Float16)kreg[i].y,
                     (_Float16)kreg[i].z, (_Float16)kreg[i].w};
            *(h4*)&lds_k[srow[i] * 128 + scol[i]] = hh;
        }
        if (kc + 1 < NCH) {                 // pipeline: next chunk in flight
            #pragma unroll
            for (int i = 0; i < 4; ++i)
                kreg[i] = *(const float4*)(kg + ((size_t)bb * NK + kr0 + 32 + srow[i]) * DD
                                           + scol[i]);
        }
        __syncthreads();

        #pragma unroll
        for (int t = 0; t < 2; ++t) {
            v8f c = {};
            #pragma unroll
            for (int kkc = 0; kkc < 4; ++kkc) {
                // B 32x16 f16 frag (same k order as A here): lane (ln,g): col = ln, halves j -> K = g*16 + j
                v16h bk = *(const v16h*)&lds_k[(t * 16 + ln) * 128 + kkc * 32 + g * 16];
                c = __builtin_amdgcn_wmma_f32_16x16x32_f16(false, aq[kkc], false, bk,
                                                           (short)0, c, false, false);
            }
            const int col = kr0 + t * 16 + ln;
            const unsigned char* mrow = maskg + (size_t)bb * NQ * NK + col;
            #pragma unroll
            for (int r = 0; r < 8; ++r) {
                const int row = qrow0 + r + 8 * g;
                float s  = mrow[(size_t)row * NK] ? NEG_BIG2 : c[r] * SC2;
                float nm = fmaxf(m[r], s);
                l[r] = l[r] * __builtin_amdgcn_exp2f(m[r] - nm)
                     + __builtin_amdgcn_exp2f(s - nm);
                m[r] = nm;
            }
        }
    }

    // combine per-lane partials across the 16-lane half-wave group
    #pragma unroll
    for (int off = 1; off < 16; off <<= 1) {
        #pragma unroll
        for (int r = 0; r < 8; ++r) {
            float mo = __shfl_xor(m[r], off, 32);
            float lo = __shfl_xor(l[r], off, 32);
            float nm = fmaxf(m[r], mo);
            l[r] = l[r] * __builtin_amdgcn_exp2f(m[r] - nm)
                 + lo * __builtin_amdgcn_exp2f(mo - nm);
            m[r] = nm;
        }
    }
    float rl[8];
    #pragma unroll
    for (int r = 0; r < 8; ++r) rl[r] = 1.0f / l[r];

    // ================= PASS 2: write att, accumulate context ===============
    v8f o[8] = {};   // 16 x 128 context accumulator (C layout, 8 d-tiles)

    float4 vreg[4];
    #pragma unroll
    for (int i = 0; i < 4; ++i) {           // preload chunk 0 (K and V)
        kreg[i] = *(const float4*)(kg + ((size_t)bb * NK + srow[i]) * DD + scol[i]);
        vreg[i] = *(const float4*)(vg + ((size_t)bb * NK + srow[i]) * DD + scol[i]);
    }

    for (int kc = 0; kc < NCH; ++kc) {
        const int kr0 = kc * 32;
        __syncthreads();
        #pragma unroll
        for (int i = 0; i < 4; ++i) {
            h4 hh = {(_Float16)kreg[i].x, (_Float16)kreg[i].y,
                     (_Float16)kreg[i].z, (_Float16)kreg[i].w};
            *(h4*)&lds_k[srow[i] * 128 + scol[i]] = hh;
            lds_vt[(scol[i] + 0) * 32 + srow[i]] = (_Float16)vreg[i].x;
            lds_vt[(scol[i] + 1) * 32 + srow[i]] = (_Float16)vreg[i].y;
            lds_vt[(scol[i] + 2) * 32 + srow[i]] = (_Float16)vreg[i].z;
            lds_vt[(scol[i] + 3) * 32 + srow[i]] = (_Float16)vreg[i].w;
        }
        if (kc + 1 < NCH) {
            #pragma unroll
            for (int i = 0; i < 4; ++i) {
                kreg[i] = *(const float4*)(kg + ((size_t)bb * NK + kr0 + 32 + srow[i]) * DD
                                           + scol[i]);
                vreg[i] = *(const float4*)(vg + ((size_t)bb * NK + kr0 + 32 + srow[i]) * DD
                                           + scol[i]);
            }
        }
        __syncthreads();

        _Float16* pb = &lds_p[w * 512];     // this wave's 16x32 P tile
        #pragma unroll
        for (int t = 0; t < 2; ++t) {
            v8f c = {};
            #pragma unroll
            for (int kkc = 0; kkc < 4; ++kkc) {
                v16h bk = *(const v16h*)&lds_k[(t * 16 + ln) * 128 + kkc * 32 + g * 16];
                c = __builtin_amdgcn_wmma_f32_16x16x32_f16(false, aq[kkc], false, bk,
                                                           (short)0, c, false, false);
            }
            const int col = kr0 + t * 16 + ln;
            #pragma unroll
            for (int r = 0; r < 8; ++r) {
                const int row = qrow0 + r + 8 * g;
                float s = maskg[((size_t)bb * NQ + row) * NK + col] ? NEG_BIG2
                                                                    : c[r] * SC2;
                float p = __builtin_amdgcn_exp2f(s - m[r]);
                att[((size_t)bb * NQ + row) * NK + col] = p * rl[r];
                pb[(r + 8 * g) * 32 + t * 16 + ln] = (_Float16)p;   // C -> A relayout
            }
        }
        // order this wave's P stores before its P loads (same-wave LDS in-order,
        // plus compile-time barrier)
        asm volatile("s_wait_dscnt 0" ::: "memory");

        v16h ap;
        {
            h8 lo = *(const h8*)&pb[ln * 32 + g * 16];
            h8 hi = *(const h8*)&pb[ln * 32 + g * 16 + 8];
            ap = __builtin_shufflevector(lo, hi, 0, 1, 2, 3, 4, 5, 6, 7,
                                         8, 9, 10, 11, 12, 13, 14, 15);
        }
        #pragma unroll
        for (int nt = 0; nt < 8; ++nt) {
            // V B-frag: col = d = nt*16+ln, K (krow) = g*16 + j, contiguous in lds_vt
            v16h bv = *(const v16h*)&lds_vt[(nt * 16 + ln) * 32 + g * 16];
            o[nt] = __builtin_amdgcn_wmma_f32_16x16x32_f16(false, ap, false, bv,
                                                           (short)0, o[nt], false, false);
        }
    }

    // ---------------- context store (normalized) ---------------------------
    #pragma unroll
    for (int nt = 0; nt < 8; ++nt) {
        #pragma unroll
        for (int r = 0; r < 8; ++r) {
            const int row = qrow0 + r + 8 * g;
            ctx[((size_t)bb * NQ + row) * DD + nt * 16 + ln] = o[nt][r] * rl[r];
        }
    }
}

extern "C" void kernel_launch(void* const* d_in, const int* in_sizes, int n_in,
                              void* d_out, int out_size, void* d_ws, size_t ws_size,
                              hipStream_t stream) {
    const float*         q    = (const float*)d_in[0];
    const float*         k    = (const float*)d_in[1];
    const float*         v    = (const float*)d_in[2];
    const unsigned char* mask = (const unsigned char*)d_in[3];

    float* ctx = (float*)d_out;                       // [B, NQ, D]
    float* att = ctx + (size_t)B_ * NQ * DD;          // [B, NQ, NK]

    dim3 grid(NQ / 128, B_);
    attn_fwd_kernel<<<grid, 256, 0, stream>>>(q, k, v, mask, ctx, att);
}
